// MultiFrameAttention_34514357191009
// MI455X (gfx1250) — matrix operand order repaired and checked by reading, not compiled, not run on hardware
//
#include <hip/hip_runtime.h>
#include <math.h>

// ---------------------------------------------------------------------------
// Types for CDNA5 WMMA (wave32)
// ---------------------------------------------------------------------------
typedef __attribute__((ext_vector_type(16))) __bf16 v16bf;
typedef __attribute__((ext_vector_type(8)))  __bf16 bf16x8;
typedef __attribute__((ext_vector_type(8)))  float  v8f;

// Problem constants
constexpr int CB  = 2;
constexpr int CS  = 1024;
constexpr int CD  = 768;
constexpr int CH  = 12;
constexpr int CF  = 6;
constexpr int CDH = 64;
constexpr int CM  = CB * CS;        // 2048 rows
constexpr int CFD = CF * CD;        // 4608
constexpr int C2D = 2 * CD;         // 1536
constexpr int CG1 = CF * CD / 4;    // 1152

// ---------------------------------------------------------------------------
// Helpers
// ---------------------------------------------------------------------------
static __device__ __forceinline__ __bf16 f2bf(float x) {
    unsigned u = __builtin_bit_cast(unsigned, x);
    unsigned r = u + 0x7fffu + ((u >> 16) & 1u);     // round-to-nearest-even
    unsigned short h = (unsigned short)(r >> 16);
    return __builtin_bit_cast(__bf16, h);
}

static __device__ __forceinline__ float gelu_exact(float x) {
    return 0.5f * x * (1.0f + erff(x * 0.70710678118654752f));
}

static __device__ __forceinline__ v8f vzero8() {
    v8f z;
#pragma unroll
    for (int i = 0; i < 8; ++i) z[i] = 0.0f;
    return z;
}

// A-fragment (16x32 bf16, MxK): lane<16 -> row=lane, elems 0..7 = K(8*half..),
// elems 8..15 = K(16+8*half..). Two contiguous 16B chunks per lane.
static __device__ __forceinline__ v16bf load_frag_a(const __bf16* base, int stride, int lane) {
    const int m    = lane & 15;
    const int half = lane >> 4;
    const bf16x8 lo = *(const bf16x8*)(base + (size_t)m * stride + 8 * half);
    const bf16x8 hi = *(const bf16x8*)(base + (size_t)m * stride + 16 + 8 * half);
    return __builtin_shufflevector(lo, hi, 0, 1, 2, 3, 4, 5, 6, 7, 8, 9, 10, 11, 12, 13, 14, 15);
}

// B-fragment (32x16 bf16, KxN) read from an N-major (transposed) buffer:
// lane holds column n = lane&15, elems 0..7 = K(8*half..), elems 8..15 = K(16+8*half..).
static __device__ __forceinline__ v16bf load_frag_b(const __bf16* baseT, int stride, int lane) {
    const int n    = lane & 15;
    const int half = lane >> 4;
    const bf16x8 lo = *(const bf16x8*)(baseT + (size_t)n * stride + 8 * half);
    const bf16x8 hi = *(const bf16x8*)(baseT + (size_t)n * stride + 16 + 8 * half);
    return __builtin_shufflevector(lo, hi, 0, 1, 2, 3, 4, 5, 6, 7, 8, 9, 10, 11, 12, 13, 14, 15);
}

static __device__ __forceinline__ v8f wmma_bf16(v16bf a, v16bf b, v8f c) {
    return __builtin_amdgcn_wmma_f32_16x16x32_bf16(false, a, false, b, (short)0, c, false, false);
}

// ---------------------------------------------------------------------------
// Generic bf16 GEMM: C[M,N] = A[M,K] @ B[K,N] + bias[N]   (f32 accumulate)
// B is supplied PRE-TRANSPOSED as BT[N][K], so both A and B tiles stage into
// LDS with 16-byte ds_store_b128 only. Block tile 128x128, 8 waves of 32x64
// (8 WMMA / wave / K-step), K-step 32, LDS double buffered -> one barrier per
// K-step. blockIdx.z = batch.
// ---------------------------------------------------------------------------
__global__ __launch_bounds__(256) void gemm_bf16_kernel(
    const __bf16* __restrict__ A, const __bf16* __restrict__ BT,
    const float* __restrict__ bias, float* __restrict__ C,
    int M, int N, int K, long long sB, long long sC, long long sBias)
{
    __shared__ __align__(16) __bf16 As[2][128][40];   // [buf][m][k]
    __shared__ __align__(16) __bf16 Bs[2][128][40];   // [buf][n][k]

    const int tid  = threadIdx.x;
    const int z    = blockIdx.z;
    const __bf16* Ab = A  + (size_t)blockIdx.y * 128 * K;
    const __bf16* Bb = BT + (size_t)z * sB + (size_t)blockIdx.x * 128 * K;

    const int row  = tid >> 1, col = (tid & 1) * 16;    // stage: 128 rows x 32 cols
    const int w    = tid >> 5, lane = tid & 31;
    const int wm   = (w >> 1) * 32, wn = (w & 1) * 64;

    v8f acc[2][4];
#pragma unroll
    for (int i = 0; i < 2; ++i)
#pragma unroll
        for (int j = 0; j < 4; ++j) acc[i][j] = vzero8();

    const int nsteps = K >> 5;

    // Prologue: stage tile 0
    {
        const bf16x8* ap = (const bf16x8*)(Ab + (size_t)row * K + col);
        bf16x8 a0 = ap[0], a1 = ap[1];
        const bf16x8* bp = (const bf16x8*)(Bb + (size_t)row * K + col);
        bf16x8 b0 = bp[0], b1 = bp[1];
        ((bf16x8*)&As[0][row][col])[0] = a0;
        ((bf16x8*)&As[0][row][col])[1] = a1;
        ((bf16x8*)&Bs[0][row][col])[0] = b0;
        ((bf16x8*)&Bs[0][row][col])[1] = b1;
    }
    __syncthreads();

    for (int step = 0; step < nsteps; ++step) {
        const int cur = step & 1;
        const bool have_next = (step + 1 < nsteps);

        // Issue next tile's global loads first; the s_wait lands after the WMMAs.
        bf16x8 na0, na1, nb0, nb1;
        if (have_next) {
            const int kt = (step + 1) << 5;
            const bf16x8* ap = (const bf16x8*)(Ab + (size_t)row * K + kt + col);
            na0 = ap[0]; na1 = ap[1];
            const bf16x8* bp = (const bf16x8*)(Bb + (size_t)row * K + kt + col);
            nb0 = bp[0]; nb1 = bp[1];
            if (step + 2 < nsteps) {
                __builtin_prefetch((const void*)(Ab + (size_t)row * K + kt + 32 + col), 0, 1);
                __builtin_prefetch((const void*)(Bb + (size_t)row * K + kt + 32 + col), 0, 1);
            }
        }

        // Compute on current buffer: 2 A frags x 4 B frags = 8 WMMA
        v16bf af[2], bfr[4];
#pragma unroll
        for (int ma = 0; ma < 2; ++ma)
            af[ma] = load_frag_a(&As[cur][wm + ma * 16][0], 40, lane);
#pragma unroll
        for (int nb2 = 0; nb2 < 4; ++nb2)
            bfr[nb2] = load_frag_b(&Bs[cur][wn + nb2 * 16][0], 40, lane);
#pragma unroll
        for (int ma = 0; ma < 2; ++ma)
#pragma unroll
            for (int nb2 = 0; nb2 < 4; ++nb2)
                acc[ma][nb2] = wmma_bf16(af[ma], bfr[nb2], acc[ma][nb2]);

        // Stage next tile into the other buffer (its readers synced last iter).
        if (have_next) {
            const int nxt = cur ^ 1;
            ((bf16x8*)&As[nxt][row][col])[0] = na0;
            ((bf16x8*)&As[nxt][row][col])[1] = na1;
            ((bf16x8*)&Bs[nxt][row][col])[0] = nb0;
            ((bf16x8*)&Bs[nxt][row][col])[1] = nb1;
            __syncthreads();
        }
    }

    const int lane16 = lane & 15, half = lane >> 4;
    const float* bz = bias + (size_t)z * sBias;
    float* Cz = C + (size_t)z * sC;
#pragma unroll
    for (int ma = 0; ma < 2; ++ma) {
#pragma unroll
        for (int nb2 = 0; nb2 < 4; ++nb2) {
            const int cn = blockIdx.x * 128 + wn + nb2 * 16 + lane16;
            const float bval = bz[cn];
#pragma unroll
            for (int r = 0; r < 8; ++r) {
                const int rm = blockIdx.y * 128 + wm + ma * 16 + r + 8 * half;
                Cz[(size_t)rm * N + cn] = acc[ma][nb2][r] + bval;
            }
        }
    }
}

// ---------------------------------------------------------------------------
// Flash attention per (b,f,h). Block = 128 threads (4 waves), each wave owns
// 16 query rows; key tiles of 64 (16 WMMA per tile per wave); online softmax.
// V is supplied pre-transposed as [B,H,DH,S] so staging is all ds_store_b128.
// Writes stacked[b,s,f*D+h*64+d] (f32 + bf16).
// ---------------------------------------------------------------------------
__global__ __launch_bounds__(128) void attn_kernel(
    const __bf16* __restrict__ qb,   // [B,F,H,S,DH]
    const __bf16* __restrict__ kb,   // [B,H,S,DH]
    const __bf16* __restrict__ vt,   // [B,H,DH,S]  (pre-transposed V)
    float* __restrict__ stacked,     // [B*S, F*D] f32
    __bf16* __restrict__ stackedb)   // [B*S, F*D] bf16
{
    __shared__ __align__(16) __bf16 Ks[64][72];        // [key][dh]
    __shared__ __align__(16) __bf16 Vs[64][72];        // [dh][key]
    __shared__ __align__(16) __bf16 Pl[4][16][72];     // per-wave P scratch [qrow][key]

    const int w = threadIdx.x >> 5, lane = threadIdx.x & 31;
    const int lane16 = lane & 15, half = lane >> 4;

    const int z  = blockIdx.y;
    const int b  = z / (CF * CH);
    const int fh = z % (CF * CH);
    const int f  = fh / CH;
    const int h  = fh % CH;

    const __bf16* Qbase = qb + ((((size_t)b * CF + f) * CH + h) * CS) * CDH;
    const __bf16* Kbase = kb + (((size_t)b * CH + h) * CS) * CDH;
    const __bf16* Vbase = vt + (((size_t)b * CH + h) * CDH) * CS;

    const int q0 = blockIdx.x * 64 + w * 16;

    // Q fragments live in registers the whole kernel (K=dh, two 32-wide chunks)
    v16bf Qf[2];
#pragma unroll
    for (int c = 0; c < 2; ++c)
        Qf[c] = load_frag_a(Qbase + (size_t)q0 * CDH + 32 * c, CDH, lane);

    float m_i[8], l_i[8];
    v8f   O[4];
#pragma unroll
    for (int r = 0; r < 8; ++r) { m_i[r] = -1e30f; l_i[r] = 0.0f; }
#pragma unroll
    for (int j = 0; j < 4; ++j) O[j] = vzero8();

    const float scale = 0.125f;               // 1/sqrt(64)
    const int srow = threadIdx.x >> 1;        // 0..63 (key row for K, dh row for V)
    const int sseg = (threadIdx.x & 1) * 32;  // 0 or 32

    for (int kt = 0; kt < CS; kt += 64) {
        __syncthreads();
        // Stage K [key][dh] and V [dh][key] — all 16B vector LDS stores
        {
            const bf16x8* kp = (const bf16x8*)(Kbase + (size_t)(kt + srow) * CDH + sseg);
            bf16x8 k0 = kp[0], k1 = kp[1], k2 = kp[2], k3 = kp[3];
            ((bf16x8*)&Ks[srow][sseg])[0] = k0;
            ((bf16x8*)&Ks[srow][sseg])[1] = k1;
            ((bf16x8*)&Ks[srow][sseg])[2] = k2;
            ((bf16x8*)&Ks[srow][sseg])[3] = k3;
            const bf16x8* vp = (const bf16x8*)(Vbase + (size_t)srow * CS + kt + sseg);
            bf16x8 v0 = vp[0], v1 = vp[1], v2 = vp[2], v3 = vp[3];
            ((bf16x8*)&Vs[srow][sseg])[0] = v0;
            ((bf16x8*)&Vs[srow][sseg])[1] = v1;
            ((bf16x8*)&Vs[srow][sseg])[2] = v2;
            ((bf16x8*)&Vs[srow][sseg])[3] = v3;
        }
        __syncthreads();

        // S = Q @ K^T : 4 key sub-tiles of 16, two dh chunks of 32 -> 8 WMMA
        v8f sf[4];
#pragma unroll
        for (int j = 0; j < 4; ++j) sf[j] = vzero8();
#pragma unroll
        for (int c = 0; c < 2; ++c) {
#pragma unroll
            for (int j = 0; j < 4; ++j) {
                v16bf kf = load_frag_b(&Ks[16 * j][32 * c], 72, lane);
                sf[j] = wmma_bf16(Qf[c], kf, sf[j]);
            }
        }

        // Online softmax update + build P in wave-private LDS
#pragma unroll
        for (int r = 0; r < 8; ++r) {
            float rmax = fmaxf(fmaxf(sf[0][r], sf[1][r]), fmaxf(sf[2][r], sf[3][r])) * scale;
#pragma unroll
            for (int off = 1; off < 16; off <<= 1)
                rmax = fmaxf(rmax, __shfl_xor(rmax, off));
            const float mnew  = fmaxf(m_i[r], rmax);
            const float alpha = __expf(m_i[r] - mnew);
            float p[4], rs = 0.0f;
#pragma unroll
            for (int j = 0; j < 4; ++j) { p[j] = __expf(sf[j][r] * scale - mnew); rs += p[j]; }
#pragma unroll
            for (int off = 1; off < 16; off <<= 1)
                rs += __shfl_xor(rs, off);
            l_i[r] = l_i[r] * alpha + rs;
            m_i[r] = mnew;
#pragma unroll
            for (int j = 0; j < 4; ++j) O[j][r] *= alpha;
            const int prow = r + 8 * half;
#pragma unroll
            for (int j = 0; j < 4; ++j)
                Pl[w][prow][16 * j + lane16] = f2bf(p[j]);
        }

        // O += P @ V : P is 16x64 -> 2 A frags; 4 dh columns of 16 -> 8 WMMA
        v16bf pf0 = load_frag_a(&Pl[w][0][0],  72, lane);
        v16bf pf1 = load_frag_a(&Pl[w][0][32], 72, lane);
#pragma unroll
        for (int j = 0; j < 4; ++j) {
            v16bf vf0 = load_frag_b(&Vs[16 * j][0],  72, lane);
            v16bf vf1 = load_frag_b(&Vs[16 * j][32], 72, lane);
            O[j] = wmma_bf16(pf0, vf0, O[j]);
            O[j] = wmma_bf16(pf1, vf1, O[j]);
        }
    }

    // Normalize and write frame-concatenated output
#pragma unroll
    for (int r = 0; r < 8; ++r) {
        const float inv = 1.0f / l_i[r];
        const int qrow = q0 + r + 8 * half;
        const size_t rowg = (size_t)b * CS + qrow;
#pragma unroll
        for (int j = 0; j < 4; ++j) {
            const float val = O[j][r] * inv;
            const size_t idx = rowg * CFD + (size_t)f * CD + h * CDH + 16 * j + lane16;
            stacked[idx]  = val;
            stackedb[idx] = f2bf(val);
        }
    }
}

// ---------------------------------------------------------------------------
// Elementwise / small kernels
// ---------------------------------------------------------------------------
__global__ void cvt_bf16_kernel(const float* __restrict__ in, __bf16* __restrict__ out, int n) {
    int i = blockIdx.x * blockDim.x + threadIdx.x;
    if (i < n) out[i] = f2bf(in[i]);
}

// Transposing convert: in [K][N] f32 -> out [N][K] bf16. blockIdx.y = batch.
__global__ void cvt_t_bf16_kernel(const float* __restrict__ in, __bf16* __restrict__ out,
                                  int K, int N) {
    const size_t nk = (size_t)N * K;
    const float* inz = in + (size_t)blockIdx.y * nk;
    __bf16* outz = out + (size_t)blockIdx.y * nk;
    int i = blockIdx.x * blockDim.x + threadIdx.x;
    if (i >= (int)nk) return;
    int n = i / K, k = i % K;
    outz[i] = f2bf(inz[(size_t)k * N + n]);
}

__global__ void gelu_cvt_kernel(const float* __restrict__ in, __bf16* __restrict__ out, int n) {
    int i = blockIdx.x * blockDim.x + threadIdx.x;
    if (i < n) out[i] = f2bf(gelu_exact(in[i]));
}

__global__ void gelu_ip_kernel(float* __restrict__ buf, int n) {
    int i = blockIdx.x * blockDim.x + threadIdx.x;
    if (i < n) buf[i] = gelu_exact(buf[i]);
}

// kvf[2048,1536] -> kb [B,H,S,DH] bf16, vt [B,H,DH,S] bf16 (V transposed)
__global__ void extract_kv_kernel(const float* __restrict__ kvf,
                                  __bf16* __restrict__ kb, __bf16* __restrict__ vt) {
    int i = blockIdx.x * blockDim.x + threadIdx.x;
    if (i >= CM * CD) return;
    int m = i / CD, d = i % CD;
    int b = m >> 10, s = m & 1023;
    int h = d >> 6,  dh = d & 63;
    kb[(((size_t)b * CH + h) * CS + s) * CDH + dh] = f2bf(kvf[(size_t)m * C2D + d]);
    vt[(((size_t)b * CH + h) * CDH + dh) * CS + s] = f2bf(kvf[(size_t)m * C2D + CD + d]);
}

// qtmp [F,2048,768] -> qb [B,F,H,S,DH] bf16
__global__ void extract_q_kernel(const float* __restrict__ qtmp, __bf16* __restrict__ qb) {
    int i = blockIdx.x * blockDim.x + threadIdx.x;
    if (i >= CF * CM * CD) return;
    int f = i / (CM * CD);
    int rem = i % (CM * CD);
    int m = rem / CD, d = rem % CD;
    int b = m >> 10, s = m & 1023;
    int h = d >> 6,  dh = d & 63;
    size_t dst = ((((size_t)b * CF + f) * CH + h) * CS + s) * CDH + dh;
    qb[dst] = f2bf(qtmp[i]);
}

// stacked += cross_scale * m2f   (becomes "mixed")
__global__ void mix_kernel(float* __restrict__ stacked, const float* __restrict__ m2f,
                           const float* __restrict__ cs, int n) {
    int i = blockIdx.x * blockDim.x + threadIdx.x;
    if (i < n) stacked[i] += cs[0] * m2f[i];
}

// gates[m, f] = softmax_f( g1f[m,:] @ W_g2 + b_g2 )
__global__ __launch_bounds__(128) void gates_kernel(
    const float* __restrict__ g1f, const float* __restrict__ Wg2,
    const float* __restrict__ bg2, float* __restrict__ gates) {
    __shared__ float red[CF][128];
    const int m = blockIdx.x, tid = threadIdx.x;
    float acc[CF];
#pragma unroll
    for (int f = 0; f < CF; ++f) acc[f] = 0.0f;
    for (int c = tid; c < CG1; c += 128) {
        const float g = g1f[(size_t)m * CG1 + c];
        const float* wr = Wg2 + (size_t)c * CF;
#pragma unroll
        for (int f = 0; f < CF; ++f) acc[f] += g * wr[f];
    }
#pragma unroll
    for (int f = 0; f < CF; ++f) red[f][tid] = acc[f];
    __syncthreads();
    for (int s = 64; s > 0; s >>= 1) {
        if (tid < s) {
#pragma unroll
            for (int f = 0; f < CF; ++f) red[f][tid] += red[f][tid + s];
        }
        __syncthreads();
    }
    if (tid == 0) {
        float v[CF], mx = -1e30f;
#pragma unroll
        for (int f = 0; f < CF; ++f) { v[f] = red[f][0] + bg2[f]; mx = fmaxf(mx, v[f]); }
        float sum = 0.0f;
#pragma unroll
        for (int f = 0; f < CF; ++f) { v[f] = __expf(v[f] - mx); sum += v[f]; }
        const float is = 1.0f / sum;
#pragma unroll
        for (int f = 0; f < CF; ++f) gates[(size_t)m * CF + f] = v[f] * is;
    }
}

// combinedb[m,d] = bf16( sum_f gates[m,f] * mixed[m, f*768+d] )
__global__ void combine_kernel(const float* __restrict__ mixed, const float* __restrict__ gates,
                               __bf16* __restrict__ combinedb) {
    int i = blockIdx.x * blockDim.x + threadIdx.x;
    if (i >= CM * CD) return;
    int m = i / CD, d = i % CD;
    const float* gr = gates + (size_t)m * CF;
    const float* mr = mixed + (size_t)m * CFD + d;
    float acc = 0.0f;
#pragma unroll
    for (int f = 0; f < CF; ++f) acc += gr[f] * mr[(size_t)f * CD];
    combinedb[i] = f2bf(acc);
}

// y = x + outf (bias already in outf); LayerNorm -> d_out
__global__ __launch_bounds__(256) void final_ln_kernel(
    const float* __restrict__ x, const float* __restrict__ outf,
    const float* __restrict__ ln_g, const float* __restrict__ ln_b,
    float* __restrict__ out) {
    __shared__ float rs[256], rs2[256];
    __shared__ float mu_s, rstd_s;
    const int m = blockIdx.x, tid = threadIdx.x;
    float yv[3], s = 0.0f, s2 = 0.0f;
#pragma unroll
    for (int i = 0; i < 3; ++i) {
        const int d = tid + i * 256;
        const float y = x[(size_t)m * CD + d] + outf[(size_t)m * CD + d];
        yv[i] = y; s += y; s2 += y * y;
    }
    rs[tid] = s; rs2[tid] = s2;
    __syncthreads();
    for (int st = 128; st > 0; st >>= 1) {
        if (tid < st) { rs[tid] += rs[tid + st]; rs2[tid] += rs2[tid + st]; }
        __syncthreads();
    }
    if (tid == 0) {
        const float mu  = rs[0] / (float)CD;
        const float var = rs2[0] / (float)CD - mu * mu;
        mu_s = mu; rstd_s = rsqrtf(var + 1e-5f);
    }
    __syncthreads();
    const float mu = mu_s, rstd = rstd_s;
#pragma unroll
    for (int i = 0; i < 3; ++i) {
        const int d = tid + i * 256;
        out[(size_t)m * CD + d] = (yv[i] - mu) * rstd * ln_g[d] + ln_b[d];
    }
}

// ---------------------------------------------------------------------------
// Host launch
// ---------------------------------------------------------------------------
static inline unsigned div_up(size_t n, unsigned b) { return (unsigned)((n + b - 1) / b); }

extern "C" void kernel_launch(void* const* d_in, const int* in_sizes, int n_in,
                              void* d_out, int out_size, void* d_ws, size_t ws_size,
                              hipStream_t stream) {
    (void)in_sizes; (void)n_in; (void)out_size; (void)ws_size;

    const float* x     = (const float*)d_in[0];
    const float* W_kv  = (const float*)d_in[1];
    const float* b_kv  = (const float*)d_in[2];
    const float* W_q   = (const float*)d_in[3];
    const float* b_q   = (const float*)d_in[4];
    const float* W_m1  = (const float*)d_in[5];
    const float* b_m1  = (const float*)d_in[6];
    const float* W_m2  = (const float*)d_in[7];
    const float* b_m2  = (const float*)d_in[8];
    const float* cs    = (const float*)d_in[9];
    const float* W_g1  = (const float*)d_in[10];
    const float* b_g1  = (const float*)d_in[11];
    const float* W_g2  = (const float*)d_in[12];
    const float* b_g2  = (const float*)d_in[13];
    const float* W_out = (const float*)d_in[14];
    const float* b_out = (const float*)d_in[15];
    const float* ln_g  = (const float*)d_in[16];
    const float* ln_b  = (const float*)d_in[17];
    float* out = (float*)d_out;

    char* base = (char*)d_ws;
    size_t off = 0;
    auto alloc = [&](size_t bytes) -> char* {
        char* p = base + off;
        off = (off + bytes + 255) & ~(size_t)255;
        return p;
    };

    __bf16* xb    = (__bf16*)alloc((size_t)CM * CD * 2);
    __bf16* wkvb  = (__bf16*)alloc((size_t)CD * C2D * 2);      // [N=1536][K=768]
    __bf16* wqb   = (__bf16*)alloc((size_t)CF * CD * CD * 2);  // [F][N=768][K=768]
    __bf16* wm1b  = (__bf16*)alloc((size_t)CFD * C2D * 2);     // [N=1536][K=4608]
    __bf16* wm2b  = (__bf16*)alloc((size_t)C2D * CFD * 2);     // [N=4608][K=1536]
    __bf16* wg1b  = (__bf16*)alloc((size_t)CD * CG1 * 2);      // [N=1152][K=768]
    __bf16* woutb = (__bf16*)alloc((size_t)CD * CD * 2);       // [N=768][K=768]
    __bf16* kb    = (__bf16*)alloc((size_t)CB * CH * CS * CDH * 2);
    __bf16* vt    = (__bf16*)alloc((size_t)CB * CH * CDH * CS * 2);
    __bf16* qb    = (__bf16*)alloc((size_t)CB * CF * CH * CS * CDH * 2);
    float*  stacked  = (float*)alloc((size_t)CM * CFD * 4);   // later becomes "mixed"
    __bf16* stackedb = (__bf16*)alloc((size_t)CM * CFD * 2);
    __bf16* h1b   = (__bf16*)alloc((size_t)CM * C2D * 2);
    float*  gates = (float*)alloc((size_t)CM * CF * 4);
    __bf16* combb = (__bf16*)alloc((size_t)CM * CD * 2);
    float*  scratch = (float*)alloc((size_t)CF * CM * CD * 4); // 37.7MB shared f32 scratch

    // 1. Converts: x plain; weights converted + transposed to [N][K]
    cvt_bf16_kernel<<<div_up((size_t)CM * CD, 256), 256, 0, stream>>>(x, xb, CM * CD);
    auto cvt_t = [&](const float* src, __bf16* dst, int K, int N, int batches) {
        cvt_t_bf16_kernel<<<dim3(div_up((size_t)N * K, 256), batches), 256, 0, stream>>>(
            src, dst, K, N);
    };
    cvt_t(W_kv,  wkvb,  CD,  C2D, 1);
    cvt_t(W_q,   wqb,   CD,  CD,  CF);
    cvt_t(W_m1,  wm1b,  CFD, C2D, 1);
    cvt_t(W_m2,  wm2b,  C2D, CFD, 1);
    cvt_t(W_g1,  wg1b,  CD,  CG1, 1);
    cvt_t(W_out, woutb, CD,  CD,  1);

    // 2. kv = x @ W_kv + b_kv  -> scratch [2048,1536]
    gemm_bf16_kernel<<<dim3(C2D / 128, CM / 128, 1), 256, 0, stream>>>(
        xb, wkvb, b_kv, scratch, CM, C2D, CD, 0, 0, 0);
    extract_kv_kernel<<<div_up((size_t)CM * CD, 256), 256, 0, stream>>>(scratch, kb, vt);

    // 3. q[f] = x @ W_q[f] + b_q[f]  -> scratch [F,2048,768]
    gemm_bf16_kernel<<<dim3(CD / 128, CM / 128, CF), 256, 0, stream>>>(
        xb, wqb, b_q, scratch, CM, CD, CD,
        (long long)CD * CD, (long long)CM * CD, (long long)CD);
    extract_q_kernel<<<div_up((size_t)CF * CM * CD, 256), 256, 0, stream>>>(scratch, qb);

    // 4. Flash attention -> stacked (f32 + bf16)
    attn_kernel<<<dim3(CS / 64, CB * CF * CH), 128, 0, stream>>>(qb, kb, vt, stacked, stackedb);

    // 5. h1 = gelu(stacked @ W_m1 + b_m1)
    gemm_bf16_kernel<<<dim3(C2D / 128, CM / 128, 1), 256, 0, stream>>>(
        stackedb, wm1b, b_m1, scratch, CM, C2D, CFD, 0, 0, 0);
    gelu_cvt_kernel<<<div_up((size_t)CM * C2D, 256), 256, 0, stream>>>(
        scratch, h1b, CM * C2D);

    // 6. mixed = stacked + cross_scale * (h1 @ W_m2 + b_m2)   (in place in `stacked`)
    gemm_bf16_kernel<<<dim3(CFD / 128, CM / 128, 1), 256, 0, stream>>>(
        h1b, wm2b, b_m2, scratch, CM, CFD, C2D, 0, 0, 0);
    mix_kernel<<<div_up((size_t)CM * CFD, 256), 256, 0, stream>>>(
        stacked, scratch, cs, CM * CFD);

    // 7. gates = softmax(gelu(x @ W_g1 + b_g1) @ W_g2 + b_g2)
    gemm_bf16_kernel<<<dim3(CG1 / 128, CM / 128, 1), 256, 0, stream>>>(
        xb, wg1b, b_g1, scratch, CM, CG1, CD, 0, 0, 0);
    gelu_ip_kernel<<<div_up((size_t)CM * CG1, 256), 256, 0, stream>>>(scratch, CM * CG1);
    gates_kernel<<<CM, 128, 0, stream>>>(scratch, W_g2, b_g2, gates);

    // 8. combined = sum_f gates * mixed_frame  (bf16)
    combine_kernel<<<div_up((size_t)CM * CD, 256), 256, 0, stream>>>(stacked, gates, combb);

    // 9. outf = combined @ W_out + b_out
    gemm_bf16_kernel<<<dim3(CD / 128, CM / 128, 1), 256, 0, stream>>>(
        combb, woutb, b_out, scratch, CM, CD, CD, 0, 0, 0);

    // 10. y = x + outf; LayerNorm -> d_out
    final_ln_kernel<<<CM, 256, 0, stream>>>(x, scratch, ln_g, ln_b, out);
}
